// TransformerEncoder_30794915512896
// MI455X (gfx1250) — matrix operand order repaired and checked by reading, not compiled, not run on hardware
//
#include <hip/hip_runtime.h>
#include <hip/hip_bf16.h>

// ---------------------------------------------------------------------------
// Transformer encoder forward (B=64, P=512, D=49, H=8, 6 blocks) -> scalar loss
// All GEMMs via v_wmma_f32_16x16x32_f16 (f16 operands, f32 accum),
// activations kept in zero-padded f16 layouts (49 -> 64).
// ---------------------------------------------------------------------------

#define BATCH      64
#define SEQ        512
#define DIM        49
#define DP         64          // padded head dim
#define NHEAD      8
#define HD         (NHEAD*DP)  // 512, padded concat width
#define NBLOCKS    6
#define INV_SCALE  (1.0f/7.000001f)   // 1/(sqrt(49)+1e-6)

typedef __attribute__((ext_vector_type(16))) _Float16 v16h;
typedef __attribute__((ext_vector_type(8)))  float    v8f;

union FragU { v16h v; unsigned u[8]; };
union HPair { _Float16 h[2]; unsigned u; };

// D = A(16x32 f16) * B(32x16 f16) + C(16x16 f32)
__device__ __forceinline__ v8f wmma_f16(v16h a, v16h b, v8f c) {
  return __builtin_amdgcn_wmma_f32_16x16x32_f16(false, a, false, b, (short)0, c,
                                                false, false);
}

// A fragment, 16x32, from row-major [m][k] f16 tile (ISA 7.12.2, 16-bit A):
// lane L -> row m = L&15; VGPR v holds k-pair k = 16*(v>>2) + 8*(L>>4) + 2*(v&3).
__device__ __forceinline__ v16h ldfrag_a(const _Float16* base, int ld) {
  const int lane = threadIdx.x & 31;
  const _Float16* row = base + (size_t)(lane & 15) * ld;
  const int hi = (lane >> 4) << 3;
  FragU f;
#pragma unroll
  for (int v = 0; v < 8; ++v) {
    const int kk = ((v >> 2) << 4) + hi + ((v & 3) << 1);
    f.u[v] = *(const unsigned*)(row + kk);   // 4B aligned: even k, even ld
  }
  return f.v;
}

// B fragment, 32x16 (KxN), sourced from a row-major [n][k] tile (i.e. W[n][k]):
// per ISA B layout, lane L -> column n = L&15; VGPR v holds k-pair
// k = 16*(v>>2) + 8*(L>>4) + 2*(v&3)  (same k order as the A fragment).
__device__ __forceinline__ v16h ldfrag_b(const _Float16* base, int ld) {
  const int lane = threadIdx.x & 31;
  const _Float16* row = base + (size_t)(lane & 15) * ld;
  const int hi = (lane >> 4) << 3;
  FragU f;
#pragma unroll
  for (int v = 0; v < 8; ++v)
    f.u[v] = *(const unsigned*)(row + ((v >> 2) << 4) + hi + ((v & 3) << 1));
  return f.v;
}

// A fragment from f32 source (LDS softmax scores); per-row scale folded in.
__device__ __forceinline__ v16h ldfrag_a_f32(const float* base, int ld, float scale) {
  const int lane = threadIdx.x & 31;
  const float* row = base + (size_t)(lane & 15) * ld;
  const int hi = (lane >> 4) << 3;
  FragU f;
#pragma unroll
  for (int v = 0; v < 8; ++v) {
    const int kk = ((v >> 2) << 4) + hi + ((v & 3) << 1);
    HPair p;
    p.h[0] = (_Float16)(row[kk]     * scale);
    p.h[1] = (_Float16)(row[kk + 1] * scale);
    f.u[v] = p.u;
  }
  return f.v;
}

// ---------------------------------------------------------------------------
// Weight / input packing to padded f16
// ---------------------------------------------------------------------------
__global__ void pack_pad_kernel(const float* __restrict__ src, _Float16* __restrict__ dst,
                                int G, int R, int C, int Rp, int Cp) {
  const long total = (long)G * Rp * Cp;
  for (long idx = blockIdx.x * (long)blockDim.x + threadIdx.x; idx < total;
       idx += (long)gridDim.x * blockDim.x) {
    const int c = (int)(idx % Cp);
    const int r = (int)((idx / Cp) % Rp);
    const int g = (int)(idx / ((long)Rp * Cp));
    float v = 0.f;
    if (r < R && c < C) v = src[((long)g * R + r) * C + c];
    dst[idx] = (_Float16)v;
  }
}

// Wout [6][49][392] -> padded [6][64][8][64]
__global__ void pack_wout_kernel(const float* __restrict__ src, _Float16* __restrict__ dst) {
  const long total = (long)NBLOCKS * DP * NHEAD * DP;
  for (long idx = blockIdx.x * (long)blockDim.x + threadIdx.x; idx < total;
       idx += (long)gridDim.x * blockDim.x) {
    const int c2 = (int)(idx % DP);
    const int h  = (int)((idx / DP) % NHEAD);
    const int o  = (int)((idx / HD) % DP);
    const int l  = (int)(idx / ((long)DP * HD));
    float v = 0.f;
    if (o < DIM && c2 < DIM) v = src[((long)l * DIM + o) * (NHEAD * DIM) + h * DIM + c2];
    dst[idx] = (_Float16)v;
  }
}

// embedding [B][P][49] -> x f16 [B][P][64]
__global__ void pack_x_kernel(const float* __restrict__ emb, _Float16* __restrict__ x) {
  const long total = (long)BATCH * SEQ * DP;
  for (long idx = blockIdx.x * (long)blockDim.x + threadIdx.x; idx < total;
       idx += (long)gridDim.x * blockDim.x) {
    const int d = (int)(idx & (DP - 1));
    const long bp = idx >> 6;
    x[idx] = (_Float16)((d < DIM) ? emb[bp * DIM + d] : 0.f);
  }
}

// ---------------------------------------------------------------------------
// Q/K/V projections: grid (P/128, H, B), 256 threads (8 waves x 16 rows).
// Q[b][h][p][64], K[b][h][p][64], Vt[b][h][64][p]  (V stored transposed)
// ---------------------------------------------------------------------------
__global__ void proj_kernel(const _Float16* __restrict__ x,
                            const _Float16* __restrict__ Wq, const _Float16* __restrict__ Wk,
                            const _Float16* __restrict__ Wv,
                            const float* __restrict__ bq, const float* __restrict__ bk,
                            const float* __restrict__ bv,
                            _Float16* __restrict__ Q, _Float16* __restrict__ K,
                            _Float16* __restrict__ Vt) {
  const int wave = threadIdx.x >> 5;
  const int lane = threadIdx.x & 31;
  const int m0 = blockIdx.x * 128 + wave * 16;
  const int h = blockIdx.y, b = blockIdx.z;
  const size_t bh = (size_t)b * NHEAD + h;

  const _Float16* xrow = x + ((size_t)b * SEQ + m0) * DP;
  const v16h a0 = ldfrag_a(xrow, DP);
  const v16h a1 = ldfrag_a(xrow + 32, DP);

  const _Float16* Ws[3] = {Wq + (size_t)h * DP * DP, Wk + (size_t)h * DP * DP,
                           Wv + (size_t)h * DP * DP};
  const float* bs[3] = {bq + h * DIM, bk + h * DIM, bv + h * DIM};
  const int n_lo = lane & 15;
  const int mrow = (lane >> 4) << 3;

#pragma unroll
  for (int t = 0; t < 3; ++t) {
#pragma unroll
    for (int nt = 0; nt < 4; ++nt) {
      const _Float16* wb = Ws[t] + (size_t)nt * 16 * DP;
      v8f c = {};
      c = wmma_f16(a0, ldfrag_b(wb, DP), c);
      c = wmma_f16(a1, ldfrag_b(wb + 32, DP), c);
      const int n = nt * 16 + n_lo;
      const float bias = (n < DIM) ? bs[t][n] : 0.f;
      if (t < 2) {
        _Float16* out = (t == 0 ? Q : K) + (bh * SEQ + m0) * DP;
#pragma unroll
        for (int j = 0; j < 8; ++j)
          out[(size_t)(mrow + j) * DP + n] = (_Float16)(c[j] + bias);
      } else {
        _Float16* out = Vt + (bh * DP + n) * (size_t)SEQ + m0;
#pragma unroll
        for (int j = 0; j < 8; ++j) out[mrow + j] = (_Float16)(c[j] + bias);
      }
    }
  }
}

// ---------------------------------------------------------------------------
// Attention: scores = QK^T/scale, clip, softmax, attn = W @ V.
// grid (P/64, H, B), 128 threads (4 waves x 16 rows). Dynamic LDS:
// per-wave 16x512 f32 score strip (32KB).
// ---------------------------------------------------------------------------
#define ATTN_WAVES 4
__global__ void attn_kernel(const _Float16* __restrict__ Q, const _Float16* __restrict__ K,
                            const _Float16* __restrict__ Vt, _Float16* __restrict__ attnb) {
  extern __shared__ float smem[];
  const int wave = threadIdx.x >> 5;
  const int lane = threadIdx.x & 31;
  const int m0 = blockIdx.x * (ATTN_WAVES * 16) + wave * 16;
  const int h = blockIdx.y, b = blockIdx.z;
  const size_t bh = (size_t)b * NHEAD + h;

  float* srow = smem + (size_t)wave * 16 * SEQ;

  const _Float16* qbase = Q + (bh * SEQ + m0) * DP;
  const v16h a0 = ldfrag_a(qbase, DP);
  const v16h a1 = ldfrag_a(qbase + 32, DP);
  const int n_lo = lane & 15;
  const int mrow = (lane >> 4) << 3;

  // score strip 16 x 512 (clip + 1/scale fused)
  for (int qt = 0; qt < SEQ / 16; ++qt) {
    const _Float16* kbase = K + (bh * SEQ + qt * 16) * DP;
    v8f c = {};
    c = wmma_f16(a0, ldfrag_b(kbase, DP), c);
    c = wmma_f16(a1, ldfrag_b(kbase + 32, DP), c);
    const int n = qt * 16 + n_lo;
#pragma unroll
    for (int j = 0; j < 8; ++j) {
      float s = c[j] * INV_SCALE;
      s = fminf(30.f, fmaxf(-30.f, s));
      srow[(size_t)(mrow + j) * SEQ + n] = s;
    }
  }
  __syncthreads();

  // softmax: 32 lanes, each owns half a row (row = lane&15, half = lane>>4);
  // half-row partials combined across the wave with shfl_xor(16).
  float rscale;
  {
    float4* r4 = (float4*)(srow + (size_t)(lane & 15) * SEQ + ((lane >> 4) << 8));
    float mx = -1e30f;
#pragma unroll 4
    for (int i = 0; i < 64; ++i) {
      const float4 v = r4[i];
      mx = fmaxf(mx, fmaxf(fmaxf(v.x, v.y), fmaxf(v.z, v.w)));
    }
    mx = fmaxf(mx, __shfl_xor(mx, 16, 32));
    float s = 0.f;
#pragma unroll 4
    for (int i = 0; i < 64; ++i) {
      float4 v = r4[i];
      v.x = __expf(v.x - mx); v.y = __expf(v.y - mx);
      v.z = __expf(v.z - mx); v.w = __expf(v.w - mx);
      s += (v.x + v.y) + (v.z + v.w);
      r4[i] = v;
    }
    s += __shfl_xor(s, 16, 32);
    rscale = 1.f / s;   // full-row reciprocal sum; row matches A-frag row (lane&15)
  }
  __syncthreads();

  // attn = softmax(scores) @ V   (A from LDS f32 with 1/sum folded, B from Vt)
  v8f acc[4] = {};
  for (int kt = 0; kt < SEQ / 32; ++kt) {
    const v16h a = ldfrag_a_f32(srow + kt * 32, SEQ, rscale);
    const _Float16* vbase = Vt + bh * DP * (size_t)SEQ + kt * 32;
#pragma unroll
    for (int nt = 0; nt < 4; ++nt)
      acc[nt] = wmma_f16(a, ldfrag_b(vbase + (size_t)nt * 16 * SEQ, SEQ), acc[nt]);
  }
  _Float16* out = attnb + (bh * SEQ + m0) * DP;
#pragma unroll
  for (int nt = 0; nt < 4; ++nt) {
    const int n = nt * 16 + n_lo;
#pragma unroll
    for (int j = 0; j < 8; ++j)
      out[(size_t)(mrow + j) * DP + n] = (_Float16)acc[nt][j];
  }
}

// ---------------------------------------------------------------------------
// head_out = attn @ Wh^T + bh, written into padded concat [b][p][h][64]
// grid (P/128, H, B), 256 threads.
// ---------------------------------------------------------------------------
__global__ void headout_kernel(const _Float16* __restrict__ attnb,
                               const _Float16* __restrict__ Wh, const float* __restrict__ bh_,
                               _Float16* __restrict__ concat) {
  const int wave = threadIdx.x >> 5;
  const int lane = threadIdx.x & 31;
  const int m0 = blockIdx.x * 128 + wave * 16;
  const int h = blockIdx.y, b = blockIdx.z;
  const size_t bh = (size_t)b * NHEAD + h;

  const _Float16* arow = attnb + (bh * SEQ + m0) * DP;
  const v16h a0 = ldfrag_a(arow, DP);
  const v16h a1 = ldfrag_a(arow + 32, DP);
  const int n_lo = lane & 15;
  const int mrow = (lane >> 4) << 3;

#pragma unroll
  for (int nt = 0; nt < 4; ++nt) {
    const _Float16* wb = Wh + (size_t)h * DP * DP + (size_t)nt * 16 * DP;
    v8f c = {};
    c = wmma_f16(a0, ldfrag_b(wb, DP), c);
    c = wmma_f16(a1, ldfrag_b(wb + 32, DP), c);
    const int n = nt * 16 + n_lo;
    const float bias = (n < DIM) ? bh_[h * DIM + n] : 0.f;
#pragma unroll
    for (int j = 0; j < 8; ++j)
      concat[((size_t)b * SEQ + m0 + mrow + j) * HD + h * DP + n] =
          (_Float16)(c[j] + bias);
  }
}

// ---------------------------------------------------------------------------
// x_next = concat @ Wout^T  (K = 512), output padded f16 [b][p][64]
// grid (P/128, 1, B), 256 threads.
// ---------------------------------------------------------------------------
__global__ void outproj_kernel(const _Float16* __restrict__ concat,
                               const _Float16* __restrict__ Wo, _Float16* __restrict__ xn) {
  const int wave = threadIdx.x >> 5;
  const int lane = threadIdx.x & 31;
  const int m0 = blockIdx.x * 128 + wave * 16;
  const int b = blockIdx.z;
  const int n_lo = lane & 15;
  const int mrow = (lane >> 4) << 3;

  const _Float16* arow = concat + ((size_t)b * SEQ + m0) * HD;
  v8f acc[4] = {};
  for (int kt = 0; kt < HD / 32; ++kt) {
    const v16h a = ldfrag_a(arow + kt * 32, HD);
#pragma unroll
    for (int nt = 0; nt < 4; ++nt)
      acc[nt] = wmma_f16(a, ldfrag_b(Wo + (size_t)nt * 16 * HD + kt * 32, HD), acc[nt]);
  }
#pragma unroll
  for (int nt = 0; nt < 4; ++nt) {
    const int n = nt * 16 + n_lo;
#pragma unroll
    for (int j = 0; j < 8; ++j)
      xn[((size_t)b * SEQ + m0 + mrow + j) * DP + n] = (_Float16)acc[nt][j];
  }
}

// ---------------------------------------------------------------------------
// mean-pool over P, MLP 49->25(relu)->10, per-batch NLL. One block per b.
// ---------------------------------------------------------------------------
__global__ void pool_mlp_kernel(const _Float16* __restrict__ x,
                                const float* __restrict__ W1, const float* __restrict__ b1,
                                const float* __restrict__ W2, const float* __restrict__ b2,
                                const int* __restrict__ labels, float* __restrict__ lossbuf) {
  __shared__ float part[4][DP];
  __shared__ float pooled[DP];
  __shared__ float h1[25];
  __shared__ float logits[10];
  const int b = blockIdx.x, t = threadIdx.x;

  // 256 threads: dim = t&63, chunk = t>>6 (4 chunks of 128 positions)
  {
    const int d = t & 63, ch = t >> 6;
    const _Float16* xb = x + ((size_t)b * SEQ + ch * 128) * DP + d;
    float s = 0.f;
    for (int p = 0; p < 128; ++p) s += (float)xb[(size_t)p * DP];
    part[ch][d] = s;
  }
  __syncthreads();
  if (t < DP)
    pooled[t] = (part[0][t] + part[1][t] + part[2][t] + part[3][t]) / (float)SEQ;
  __syncthreads();
  if (t < 25) {
    float s = b1[t];
    for (int d = 0; d < DIM; ++d) s += W1[t * DIM + d] * pooled[d];
    h1[t] = fmaxf(s, 0.f);
  }
  __syncthreads();
  if (t < 10) {
    float s = b2[t];
    for (int j = 0; j < 25; ++j) s += W2[t * 25 + j] * h1[j];
    logits[t] = s;
  }
  __syncthreads();
  if (t == 0) {
    float mx = logits[0];
    for (int o = 1; o < 10; ++o) mx = fmaxf(mx, logits[o]);
    float se = 0.f;
    for (int o = 0; o < 10; ++o) se += __expf(logits[o] - mx);
    const float lse = mx + __logf(se);
    const int y = labels[b];
    lossbuf[b] = -(logits[y] - lse);
  }
}

__global__ void reduce_loss_kernel(const float* __restrict__ lossbuf, float* __restrict__ out) {
  if (threadIdx.x == 0 && blockIdx.x == 0) {
    float s = 0.f;
    for (int i = 0; i < BATCH; ++i) s += lossbuf[i];
    out[0] = s / (float)BATCH;
  }
}

// ---------------------------------------------------------------------------
// Host launcher
// ---------------------------------------------------------------------------
extern "C" void kernel_launch(void* const* d_in, const int* in_sizes, int n_in,
                              void* d_out, int out_size, void* d_ws, size_t ws_size,
                              hipStream_t stream) {
  const float* emb    = (const float*)d_in[0];
  const int*   labels = (const int*)d_in[1];
  const float* Wq = (const float*)d_in[2];  const float* bq = (const float*)d_in[3];
  const float* Wk = (const float*)d_in[4];  const float* bk = (const float*)d_in[5];
  const float* Wv = (const float*)d_in[6];  const float* bv = (const float*)d_in[7];
  const float* Wh = (const float*)d_in[8];  const float* bh = (const float*)d_in[9];
  const float* Wout = (const float*)d_in[10];
  const float* W1 = (const float*)d_in[11]; const float* b1 = (const float*)d_in[12];
  const float* W2 = (const float*)d_in[13]; const float* b2 = (const float*)d_in[14];

  char* ws = (char*)d_ws;
  size_t off = 0;
  auto take = [&](size_t bytes) -> char* {
    char* p = ws + off;
    off += (bytes + 255) & ~(size_t)255;
    return p;
  };

  const size_t WQKV_HALVES = (size_t)NBLOCKS * NHEAD * DP * DP;  // 196608
  _Float16* WqP  = (_Float16*)take(WQKV_HALVES * 2);
  _Float16* WkP  = (_Float16*)take(WQKV_HALVES * 2);
  _Float16* WvP  = (_Float16*)take(WQKV_HALVES * 2);
  _Float16* WhP  = (_Float16*)take(WQKV_HALVES * 2);
  _Float16* WoP  = (_Float16*)take((size_t)NBLOCKS * DP * HD * 2);
  _Float16* xA   = (_Float16*)take((size_t)BATCH * SEQ * DP * 2);
  _Float16* xB   = (_Float16*)take((size_t)BATCH * SEQ * DP * 2);
  _Float16* Qb   = (_Float16*)take((size_t)BATCH * NHEAD * SEQ * DP * 2);
  _Float16* Kb   = (_Float16*)take((size_t)BATCH * NHEAD * SEQ * DP * 2);
  _Float16* Vt   = (_Float16*)take((size_t)BATCH * NHEAD * DP * SEQ * 2);
  _Float16* Ab   = (_Float16*)take((size_t)BATCH * NHEAD * SEQ * DP * 2);
  _Float16* Cc   = (_Float16*)take((size_t)BATCH * SEQ * HD * 2);
  float*    lossbuf = (float*)take(BATCH * sizeof(float));

  // pack weights + input
  pack_pad_kernel<<<768, 256, 0, stream>>>(Wq, WqP, NBLOCKS * NHEAD, DIM, DIM, DP, DP);
  pack_pad_kernel<<<768, 256, 0, stream>>>(Wk, WkP, NBLOCKS * NHEAD, DIM, DIM, DP, DP);
  pack_pad_kernel<<<768, 256, 0, stream>>>(Wv, WvP, NBLOCKS * NHEAD, DIM, DIM, DP, DP);
  pack_pad_kernel<<<768, 256, 0, stream>>>(Wh, WhP, NBLOCKS * NHEAD, DIM, DIM, DP, DP);
  pack_wout_kernel<<<768, 256, 0, stream>>>(Wout, WoP);
  pack_x_kernel<<<4096, 256, 0, stream>>>(emb, xA);

  _Float16* cur = xA;
  _Float16* nxt = xB;
  const size_t attn_smem = (size_t)ATTN_WAVES * 16 * SEQ * sizeof(float);

  for (int l = 0; l < NBLOCKS; ++l) {
    const size_t wo = (size_t)l * NHEAD * DP * DP;
    const size_t bo = (size_t)l * NHEAD * DIM;
    proj_kernel<<<dim3(SEQ / 128, NHEAD, BATCH), 256, 0, stream>>>(
        cur, WqP + wo, WkP + wo, WvP + wo, bq + bo, bk + bo, bv + bo, Qb, Kb, Vt);
    attn_kernel<<<dim3(SEQ / (ATTN_WAVES * 16), NHEAD, BATCH), ATTN_WAVES * 32,
                  attn_smem, stream>>>(Qb, Kb, Vt, Ab);
    headout_kernel<<<dim3(SEQ / 128, NHEAD, BATCH), 256, 0, stream>>>(
        Ab, WhP + wo, bh + bo, Cc);
    outproj_kernel<<<dim3(SEQ / 128, 1, BATCH), 256, 0, stream>>>(
        Cc, WoP + (size_t)l * DP * HD, nxt);
    _Float16* t = cur; cur = nxt; nxt = t;
  }

  pool_mlp_kernel<<<BATCH, 256, 0, stream>>>(cur, W1, b1, W2, b2, labels, lossbuf);
  reduce_loss_kernel<<<1, 32, 0, stream>>>(lossbuf, (float*)d_out);
}
